// MultiHeadAttention_69449621176705
// MI455X (gfx1250) — matrix operand order repaired and checked by reading, not compiled, not run on hardware
//
#include <hip/hip_runtime.h>
#include <hip/hip_bf16.h>

typedef __attribute__((ext_vector_type(16))) _Float16 v16h;
typedef __attribute__((ext_vector_type(8)))  _Float16 v8h;
typedef __attribute__((ext_vector_type(8)))  float    v8f;
typedef __attribute__((ext_vector_type(4)))  unsigned u32x4;
typedef __attribute__((ext_vector_type(8)))  unsigned u32x8;

#define D_MODEL 1024
#define HEADS   16
#define DK      64
#define SEQ     2048
#define BATCH   2
#define MROWS   4096            // B*SEQ
#define NEGINF  (-1.0e9f)

// Build a v16h fragment from two 8-half chunks (each 16B-aligned in our layouts).
__device__ inline v16h make_frag(const _Float16* lo, const _Float16* hi) {
  v8h a = *(const v8h*)lo;
  v8h b = *(const v8h*)hi;
  return __builtin_shufflevector(a, b, 0,1,2,3,4,5,6,7,8,9,10,11,12,13,14,15);
}

#define LDS_STRIDE 72   // 64 + 8 halves pad: 144B row stride breaks bank conflicts

// ---------------------------------------------------------------------------
// Tensor Data Mover: 2-D f16 tile (tile1 rows x tile0 halves, row stride
// stride0 halves) from global -> LDS, with HW padding of +4 DWORDs every
// 32 DWORDs (=> 72-half LDS row stride for tile0==64).
// D# per CDNA5 ISA; 2-D tensor -> groups 0/1 only.
// ---------------------------------------------------------------------------
__device__ inline void tdm_load_2d_f16(unsigned lds_byte_addr,
                                       const _Float16* gptr,
                                       unsigned tile0, unsigned tile1,
                                       unsigned long long stride0)
{
  unsigned long long ga = (unsigned long long)(uintptr_t)gptr;
  u32x4 g0;
  g0[0] = 1u;                                            // count=1 (valid, user)
  g0[1] = lds_byte_addr;                                 // lds_addr
  g0[2] = (unsigned)(ga & 0xFFFFFFFFu);                  // global_addr[31:0]
  g0[3] = (unsigned)((ga >> 32) & 0x01FFFFFFu) | (2u << 30);  // addr[56:32], type=2
  u32x8 g1;
  g1[0] = (1u << 16)        // data_size = 2 bytes
        | (1u << 20)        // pad_enable
        | (4u << 22)        // pad_interval: 32 DWORDs (128B = one 64-half row)
        | (3u << 25);       // pad_amount: 4 DWORDs (16B = 8 halves)
  g1[1] = (tile0 & 0xFFFFu) << 16;                       // tensor_dim0[15:0]
  g1[2] = (tile0 >> 16) | ((tile1 & 0xFFFFu) << 16);     // dim0 hi | tensor_dim1 lo
  g1[3] = (tile1 >> 16) | (tile0 << 16);                 // dim1 hi | tile_dim0
  g1[4] = tile1;                                         // tile_dim1 (tile_dim2=0)
  g1[5] = (unsigned)(stride0 & 0xFFFFFFFFu);             // tensor_dim0_stride lo
  g1[6] = (unsigned)((stride0 >> 32) & 0xFFFFu);         // stride0 hi (stride1=0)
  g1[7] = 0u;
  asm volatile("tensor_load_to_lds %0, %1" :: "s"(g0), "s"(g1) : "memory");
}

// ---------------------------------------------------------------------------
// GEMM: C[M,N] = X[M,K] * W[N,K]^T + bias, M=4096, N=K=1024.
// OMODE 0: f16 out, row-major [M][N]      (Q, K projections)
// OMODE 1: f16 out, transposed [N][M]     (V projection -> Vt for attention)
// OMODE 2: f32 out, row-major [M][N]      (final projection -> d_out)
// ---------------------------------------------------------------------------
#define GK_MT 128
#define GK_NT 64
#define GK_KT 64

template<typename XT, int OMODE>
__global__ __launch_bounds__(256)
void gemm_xwT_kernel(const XT* __restrict__ X, const float* __restrict__ W,
                     const float* __restrict__ bias, void* __restrict__ outp,
                     float scale)
{
  __shared__ _Float16 As[GK_MT][LDS_STRIDE];
  __shared__ _Float16 Bs[GK_NT][LDS_STRIDE];

  const int tid  = threadIdx.x;
  const int lane = tid & 31;
  const int w    = tid >> 5;          // 0..7 waves
  const int lm   = lane & 15;
  const int hi   = lane >> 4;         // half-wave select
  const int mBase = blockIdx.y * GK_MT;
  const int nBase = blockIdx.x * GK_NT;
  const int wm = (w >> 1) * 32;       // wave M offset in tile
  const int wn = (w & 1) * 32;        // wave N offset in tile

  v8f c[2][2];
  #pragma unroll
  for (int mi = 0; mi < 2; ++mi)
    #pragma unroll
    for (int ni = 0; ni < 2; ++ni)
      #pragma unroll
      for (int e = 0; e < 8; ++e) c[mi][ni][e] = 0.0f;

  const int ar   = tid >> 1;          // 0..127 A-tile row
  const int aseg = (tid & 1) * 32;    // 0/32 halves
  const int br   = tid >> 2;          // 0..63  B-tile row
  const int bseg = (tid & 3) * 16;    // 0/16/32/48 halves

  for (int k0 = 0; k0 < D_MODEL; k0 += GK_KT) {
    // ---- stage A (convert to f16 if needed; packed b128 LDS stores) ----
    {
      const XT* src = X + (size_t)(mBase + ar) * D_MODEL + k0 + aseg;
      if constexpr (sizeof(XT) == 4) {
        #pragma unroll
        for (int j = 0; j < 4; ++j) {
          float4 f0 = *(const float4*)((const char*)src + j * 32);
          float4 f1 = *(const float4*)((const char*)src + j * 32 + 16);
          v8h t;
          t[0] = (_Float16)f0.x; t[1] = (_Float16)f0.y;
          t[2] = (_Float16)f0.z; t[3] = (_Float16)f0.w;
          t[4] = (_Float16)f1.x; t[5] = (_Float16)f1.y;
          t[6] = (_Float16)f1.z; t[7] = (_Float16)f1.w;
          *(v8h*)&As[ar][aseg + j*8] = t;
        }
      } else {
        #pragma unroll
        for (int j = 0; j < 4; ++j)
          *(v8h*)&As[ar][aseg + j*8] = *(const v8h*)((const char*)src + j * 16);
      }
    }
    // ---- stage B (weights, f32 -> f16, packed) ----
    {
      const float* src = W + (size_t)(nBase + br) * D_MODEL + k0 + bseg;
      #pragma unroll
      for (int j = 0; j < 2; ++j) {
        float4 f0 = *(const float4*)((const char*)src + j * 32);
        float4 f1 = *(const float4*)((const char*)src + j * 32 + 16);
        v8h t;
        t[0] = (_Float16)f0.x; t[1] = (_Float16)f0.y;
        t[2] = (_Float16)f0.z; t[3] = (_Float16)f0.w;
        t[4] = (_Float16)f1.x; t[5] = (_Float16)f1.y;
        t[6] = (_Float16)f1.z; t[7] = (_Float16)f1.w;
        *(v8h*)&Bs[br][bseg + j*8] = t;
      }
    }
    if (k0 + GK_KT < D_MODEL) {  // prefetch next k-tile (global_prefetch_b8)
      __builtin_prefetch((const char*)(X + (size_t)(mBase + ar) * D_MODEL + k0 + GK_KT + aseg), 0, 1);
      __builtin_prefetch(W + (size_t)(nBase + br) * D_MODEL + k0 + GK_KT + bseg, 0, 1);
    }
    __syncthreads();

    #pragma unroll
    for (int ks = 0; ks < 2; ++ks) {
      // load all fragments of this k-step first, then issue WMMAs back-to-back
      v16h a[2], b[2];
      #pragma unroll
      for (int mi = 0; mi < 2; ++mi) {
        const _Float16* p = &As[wm + mi*16 + lm][ks*32 + (hi ? 8 : 0)];
        a[mi] = make_frag(p, p + 16);          // A 16x32 layout
      }
      #pragma unroll
      for (int ni = 0; ni < 2; ++ni) {
        const _Float16* p = &Bs[wn + ni*16 + lm][ks*32 + (hi ? 8 : 0)];
        b[ni] = make_frag(p, p + 16);          // B 32x16 layout
      }
      #pragma unroll
      for (int mi = 0; mi < 2; ++mi)
        #pragma unroll
        for (int ni = 0; ni < 2; ++ni)
          c[mi][ni] = __builtin_amdgcn_wmma_f32_16x16x32_f16(
              false, a[mi], false, b[ni], (short)0, c[mi][ni], false, false);
    }
    __syncthreads();
  }

  // ---- epilogue: bias + scale, layout per OMODE ----
  #pragma unroll
  for (int mi = 0; mi < 2; ++mi) {
    #pragma unroll
    for (int ni = 0; ni < 2; ++ni) {
      const int gcol = nBase + wn + ni*16 + lm;
      const float bv = bias[gcol];
      #pragma unroll
      for (int i = 0; i < 8; ++i) {
        const int grow = mBase + wm + mi*16 + i + 8*hi;   // C/D row mapping
        float val = (c[mi][ni][i] + bv) * scale;
        if constexpr (OMODE == 0) {
          ((_Float16*)outp)[(size_t)grow * D_MODEL + gcol] = (_Float16)val;
        } else if constexpr (OMODE == 1) {
          ((_Float16*)outp)[(size_t)gcol * MROWS + grow] = (_Float16)val;
        } else {
          ((float*)outp)[(size_t)grow * D_MODEL + gcol] = val;
        }
      }
    }
  }
}

// ---------------------------------------------------------------------------
// Flash attention. Grid (Sq/64, H, B), block 128 (4 waves); each wave owns a
// 16-row Q tile; streams Sk in 64-wide blocks with online softmax.
// K / V^T tiles are staged by the Tensor Data Mover (wave 0 issues, double-
// buffered; iteration `it` waits on tile `it` while tile `it+1` streams in).
// ---------------------------------------------------------------------------
#define NIT (SEQ / 64)

__global__ __launch_bounds__(128)
void flash_attn_kernel(const _Float16* __restrict__ Qh,
                       const _Float16* __restrict__ Kh,
                       const _Float16* __restrict__ VtH,   // [H*DK][MROWS]
                       const int* __restrict__ mask,
                       _Float16* __restrict__ Xh)
{
  __shared__ _Float16 Ks[2][64][LDS_STRIDE];   // [buf][kk][d]
  __shared__ _Float16 Vs[2][64][LDS_STRIDE];   // [buf][d][kk]  (V transposed)
  __shared__ _Float16 Ps[4][16][LDS_STRIDE];   // per-wave P scratch

  const int tid  = threadIdx.x;
  const int lane = tid & 31;
  const int w    = tid >> 5;       // 0..3
  const int lm   = lane & 15;
  const int hi   = lane >> 4;
  const int b    = blockIdx.z;
  const int h    = blockIdx.y;
  const int bbase = b * SEQ;
  const int qbase = blockIdx.x * 64 + w * 16;  // local q row base of this wave

  // Q fragments (scale 1/sqrt(dk) folded into Q projection epilogue)
  v16h aq[2];
  {
    const _Float16* qp = Qh + (size_t)(bbase + qbase + lm) * D_MODEL + h * DK;
    #pragma unroll
    for (int ks = 0; ks < 2; ++ks) {
      const _Float16* p = qp + ks*32 + (hi ? 8 : 0);
      aq[ks] = make_frag(p, p + 16);
    }
  }

  v8f o[4];
  float m_run[8], l_run[8];
  #pragma unroll
  for (int t = 0; t < 4; ++t)
    #pragma unroll
    for (int e = 0; e < 8; ++e) o[t][e] = 0.0f;
  #pragma unroll
  for (int i = 0; i < 8; ++i) { m_run[i] = -3.0e38f; l_run[i] = 0.0f; }

  // TDM prologue: tile 0 into buffer 0 (wave 0 only; TENSORcnt is per-wave)
  if (w == 0) {
    tdm_load_2d_f16((unsigned)(uintptr_t)&Ks[0][0][0],
                    Kh + (size_t)bbase * D_MODEL + h * DK, 64, 64, D_MODEL);
    tdm_load_2d_f16((unsigned)(uintptr_t)&Vs[0][0][0],
                    VtH + (size_t)(h * DK) * MROWS + bbase, 64, 64, MROWS);
  }

  for (int it = 0; it < NIT; ++it) {
    const int kv0 = it * 64;
    const int buf = it & 1;
    if (w == 0) {
      if (it + 1 < NIT) {
        const int nb = (it + 1) & 1;
        tdm_load_2d_f16((unsigned)(uintptr_t)&Ks[nb][0][0],
                        Kh + (size_t)(bbase + kv0 + 64) * D_MODEL + h * DK,
                        64, 64, D_MODEL);
        tdm_load_2d_f16((unsigned)(uintptr_t)&Vs[nb][0][0],
                        VtH + (size_t)(h * DK) * MROWS + bbase + kv0 + 64,
                        64, 64, MROWS);
        __builtin_amdgcn_s_wait_tensorcnt(2);   // current tile landed
      } else {
        __builtin_amdgcn_s_wait_tensorcnt(0);
      }
    }
    __syncthreads();

    // S = Q * K^T  (16 q-rows x 64 k-cols, 8 WMMAs) — load frags, then WMMA
    v8f s[4];
    #pragma unroll
    for (int ni = 0; ni < 4; ++ni)
      #pragma unroll
      for (int e = 0; e < 8; ++e) s[ni][e] = 0.0f;
    #pragma unroll
    for (int ks = 0; ks < 2; ++ks) {
      v16h kb[4];
      #pragma unroll
      for (int ni = 0; ni < 4; ++ni) {
        const _Float16* p = &Ks[buf][ni*16 + lm][ks*32 + (hi ? 8 : 0)];
        kb[ni] = make_frag(p, p + 16);
      }
      #pragma unroll
      for (int ni = 0; ni < 4; ++ni)
        s[ni] = __builtin_amdgcn_wmma_f32_16x16x32_f16(
            false, aq[ks], false, kb[ni], (short)0, s[ni], false, false);
    }

    // mask (reference: score = mask ? score : -1e9)
    #pragma unroll
    for (int ni = 0; ni < 4; ++ni) {
      const int col = kv0 + ni*16 + lm;
      #pragma unroll
      for (int i = 0; i < 8; ++i) {
        const int qrow = qbase + i + 8*hi;
        if (mask[((size_t)b * SEQ + qrow) * SEQ + col] == 0) s[ni][i] = NEGINF;
      }
    }

    // online softmax: cross-lane row reductions within each 16-lane half
    float mb[8], rs[8], alpha[8];
    #pragma unroll
    for (int i = 0; i < 8; ++i)
      mb[i] = fmaxf(fmaxf(s[0][i], s[1][i]), fmaxf(s[2][i], s[3][i]));
    #pragma unroll
    for (int off = 8; off >= 1; off >>= 1)
      #pragma unroll
      for (int i = 0; i < 8; ++i)
        mb[i] = fmaxf(mb[i], __shfl_xor(mb[i], off, 32));
    #pragma unroll
    for (int i = 0; i < 8; ++i) {
      float mn = fmaxf(m_run[i], mb[i]);
      alpha[i] = __expf(m_run[i] - mn);
      m_run[i] = mn;
      rs[i] = 0.0f;
    }
    #pragma unroll
    for (int ni = 0; ni < 4; ++ni)
      #pragma unroll
      for (int i = 0; i < 8; ++i) {
        float p = __expf(s[ni][i] - m_run[i]);
        rs[i] += p;
        Ps[w][i + 8*hi][ni*16 + lm] = (_Float16)p;   // C/D layout -> row-major LDS
      }
    #pragma unroll
    for (int off = 8; off >= 1; off >>= 1)
      #pragma unroll
      for (int i = 0; i < 8; ++i)
        rs[i] += __shfl_xor(rs[i], off, 32);
    #pragma unroll
    for (int i = 0; i < 8; ++i) l_run[i] = l_run[i] * alpha[i] + rs[i];
    #pragma unroll
    for (int t = 0; t < 4; ++t)
      #pragma unroll
      for (int i = 0; i < 8; ++i) o[t][i] *= alpha[i];

    // wave-local LDS RAW: LDS ops are in-order per wave; wait + compiler fence
    asm volatile("s_wait_dscnt 0" ::: "memory");

    // O += P * V  (8 WMMAs) — load frags, then WMMA
    #pragma unroll
    for (int ks = 0; ks < 2; ++ks) {
      const _Float16* pp = &Ps[w][lm][ks*32 + (hi ? 8 : 0)];
      v16h pa = make_frag(pp, pp + 16);              // A-layout reload of P
      v16h vb[4];
      #pragma unroll
      for (int t = 0; t < 4; ++t) {
        const _Float16* vp = &Vs[buf][t*16 + lm][ks*32 + (hi ? 8 : 0)];
        vb[t] = make_frag(vp, vp + 16);
      }
      #pragma unroll
      for (int t = 0; t < 4; ++t)
        o[t] = __builtin_amdgcn_wmma_f32_16x16x32_f16(
            false, pa, false, vb[t], (short)0, o[t], false, false);
    }
    __syncthreads();   // readers done before TDM refills this buffer next iter
  }

  // finalize: divide by l, store f16 [B*S][D] (head-interleaved = reference reshape)
  float rinv[8];
  #pragma unroll
  for (int i = 0; i < 8; ++i) rinv[i] = 1.0f / l_run[i];
  #pragma unroll
  for (int t = 0; t < 4; ++t) {
    const int col = h*DK + t*16 + lm;
    #pragma unroll
    for (int i = 0; i < 8; ++i) {
      const int qrow = bbase + qbase + i + 8*hi;
      Xh[(size_t)qrow * D_MODEL + col] = (_Float16)(o[t][i] * rinv[i]);
    }
  }
}

// ---------------------------------------------------------------------------
extern "C" void kernel_launch(void* const* d_in, const int* in_sizes, int n_in,
                              void* d_out, int out_size, void* d_ws, size_t ws_size,
                              hipStream_t stream)
{
  const float* query = (const float*)d_in[0];
  const float* key_  = (const float*)d_in[1];
  const float* value = (const float*)d_in[2];
  const int*   mask  = (const int*)d_in[3];
  const float* W0 = (const float*)d_in[4];
  const float* b0 = (const float*)d_in[5];
  const float* W1 = (const float*)d_in[6];
  const float* b1 = (const float*)d_in[7];
  const float* W2 = (const float*)d_in[8];
  const float* b2 = (const float*)d_in[9];
  const float* W3 = (const float*)d_in[10];
  const float* b3 = (const float*)d_in[11];

  char* ws = (char*)d_ws;
  const size_t mat16 = (size_t)MROWS * D_MODEL * sizeof(_Float16);  // 8 MB
  _Float16* Qh  = (_Float16*)(ws);
  _Float16* Kh  = (_Float16*)(ws + mat16);
  _Float16* VtH = (_Float16*)(ws + 2 * mat16);
  _Float16* Xh  = (_Float16*)(ws + 3 * mat16);

  dim3 gblk(D_MODEL / GK_NT, MROWS / GK_MT);   // (16, 32)
  const float qscale = 0.125f;                 // 1/sqrt(dk), folded into Q proj

  gemm_xwT_kernel<float, 0><<<gblk, 256, 0, stream>>>(query, W0, b0, Qh,  qscale);
  gemm_xwT_kernel<float, 0><<<gblk, 256, 0, stream>>>(key_,  W1, b1, Kh,  1.0f);
  gemm_xwT_kernel<float, 1><<<gblk, 256, 0, stream>>>(value, W2, b2, VtH, 1.0f);

  dim3 ga(SEQ / 64, HEADS, BATCH);             // (32, 16, 2)
  flash_attn_kernel<<<ga, 128, 0, stream>>>(Qh, Kh, VtH, mask, Xh);

  gemm_xwT_kernel<_Float16, 2><<<gblk, 256, 0, stream>>>(Xh, W3, b3, d_out, 1.0f);
}
